// ModelWithSTMGNNLayer_84224308674633
// MI455X (gfx1250) — matrix operand order repaired and checked by reading, not compiled, not run on hardware
//
#include <hip/hip_runtime.h>

// ---------------- problem constants ----------------
constexpr int N_   = 20000;
constexpr int E_   = 320000;
constexpr int FEAT = 128;
constexpr int C_   = 256;
constexpr int H_   = 8;
constexpr int D_   = 32;
constexpr int L_   = 5;
constexpr int T_   = 64;
constexpr int MS_  = 10;
constexpr int MD_  = 128;
constexpr int NC_  = 2;
constexpr float NEG_SLOPE = 0.2f;

typedef __attribute__((ext_vector_type(16))) __bf16        v16bf;
typedef __attribute__((ext_vector_type(8)))  float         v8f;
typedef __attribute__((ext_vector_type(8)))  unsigned int  v8u;

// ---------------- fill kernels ----------------
__global__ void fill_f32(float* __restrict__ p, float v, long long n) {
    long long i = (long long)blockIdx.x * blockDim.x + threadIdx.x;
    if (i < n) p[i] = v;
}
__global__ void fill_u32(unsigned* __restrict__ p, unsigned v, long long n) {
    long long i = (long long)blockIdx.x * blockDim.x + threadIdx.x;
    if (i < n) p[i] = v;
}

// ---------------- weight pre-pack: f32 [K x Nc] -> bf16, wave-pair-interleaved --------
// dword index = ((((p*(K/32) + ks)*2 + t)*32 + lane)*8 + j), tile nt = p + 8*t
//   -> for wave p at step ks, BOTH of its tiles' fragments are contiguous (512 dwords),
//      so the GEMM loads them from one base register with immediate offsets.
//   lane: half=lane>>4 selects K {0..7,16..23} vs {8..15,24..31}; lrow=lane&15 = column within tile
//   dword j holds the bf16 pair (k = ks*32 + (j>>2)*16 + half*8 + 2*(j&3), k+1) of column nt*16+lrow
__global__ void pack_b_bf16(const float* __restrict__ B, unsigned* __restrict__ Bp,
                            int K, int Nc)
{
    long long idx = (long long)blockIdx.x * blockDim.x + threadIdx.x;
    long long total = (long long)(Nc >> 4) * (K >> 5) * 32 * 8;
    if (idx >= total) return;
    int       j    = (int)(idx & 7);
    int       lane = (int)((idx >> 3) & 31);
    int       t    = (int)((idx >> 8) & 1);
    long long rest = idx >> 9;
    int ksteps = K >> 5;
    int ks = (int)(rest % ksteps);
    int p  = (int)(rest / ksteps);
    int half = lane >> 4, lrow = lane & 15;
    int nt  = p + 8 * t;
    int col = nt * 16 + lrow;
    int k   = ks * 32 + (j >> 2) * 16 + half * 8 + 2 * (j & 3);
    unsigned short lo = __builtin_bit_cast(unsigned short, (__bf16)B[(size_t)k * Nc + col]);
    unsigned short hi = __builtin_bit_cast(unsigned short, (__bf16)B[(size_t)(k + 1) * Nc + col]);
    Bp[idx] = (unsigned)lo | ((unsigned)hi << 16);
}

// ---------------- WMMA GEMM: C = act(A[MxK] @ B[KxNc=256] + bias + res) ----------------
// One block = 16 rows; 8 waves; wave w owns column tiles w and w+8 (Nc==256 -> 16 tiles).
// Simple loop body; compiler software-pipelines the clustered b128 loads. Both B
// fragments come from one base pointer (immediate offsets 0 / +1KB) thanks to the
// pair-interleaved pack layout.
template <bool HAS_BIAS, bool HAS_RES, bool RELU>
__global__ void wmma_gemm(const float* __restrict__ A, const unsigned* __restrict__ Bp,
                          const float* __restrict__ bias, const float* __restrict__ res,
                          float* __restrict__ Cout, int M, int K, int Nc)
{
    const int lane  = threadIdx.x & 31;
    const int wave  = threadIdx.x >> 5;   // 0..7 == tile pair p
    const int lrow  = lane & 15;
    const int half  = lane >> 4;          // 0 or 1
    const int mBase = blockIdx.x * 16;
    const int ksteps = K >> 5;

    v8f acc0 = {}, acc1 = {};
    const float*    Abase = A + (size_t)(mBase + lrow) * K + half * 8;
    const unsigned* Bbase = Bp + ((size_t)wave * ksteps << 9) + ((size_t)lane << 3);

    for (int ks = 0; ks < ksteps; ++ks) {
        const float* Arow = Abase + ks * 32;
        float4 f0 = *(const float4*)(Arow);
        float4 f1 = *(const float4*)(Arow + 4);
        float4 f2 = *(const float4*)(Arow + 16);
        float4 f3 = *(const float4*)(Arow + 20);
        // speculative prefetch two steps ahead (fault-dropped; safe past buffer end)
        __builtin_prefetch(Arow + 64, 0, 0);

        const unsigned* bptr = Bbase + ((size_t)ks << 9);
        v16bf b0 = __builtin_bit_cast(v16bf, *(const v8u*)(bptr));
        v16bf b1 = __builtin_bit_cast(v16bf, *(const v8u*)(bptr + 256));

        v16bf a;
        a[0]  = (__bf16)f0.x; a[1]  = (__bf16)f0.y; a[2]  = (__bf16)f0.z; a[3]  = (__bf16)f0.w;
        a[4]  = (__bf16)f1.x; a[5]  = (__bf16)f1.y; a[6]  = (__bf16)f1.z; a[7]  = (__bf16)f1.w;
        a[8]  = (__bf16)f2.x; a[9]  = (__bf16)f2.y; a[10] = (__bf16)f2.z; a[11] = (__bf16)f2.w;
        a[12] = (__bf16)f3.x; a[13] = (__bf16)f3.y; a[14] = (__bf16)f3.z; a[15] = (__bf16)f3.w;

        acc0 = __builtin_amdgcn_wmma_f32_16x16x32_bf16(false, a, false, b0, (short)0, acc0,
                                                       false, false);
        acc1 = __builtin_amdgcn_wmma_f32_16x16x32_bf16(false, a, false, b1, (short)0, acc1,
                                                       false, false);
    }

    // D layout: lane -> column nBase+lrow; VGPR r -> row r + 8*half
    {
        const int n = wave * 16 + lrow;
        const float bv = HAS_BIAS ? bias[n] : 0.f;
        #pragma unroll
        for (int r = 0; r < 8; ++r) {
            int m = mBase + r + 8 * half;
            float v = acc0[r] + bv;
            if (HAS_RES) v += res[(size_t)m * Nc + n];
            if (RELU)    v = fmaxf(v, 0.f);
            Cout[(size_t)m * Nc + n] = v;
        }
    }
    {
        const int n = (wave + 8) * 16 + lrow;
        const float bv = HAS_BIAS ? bias[n] : 0.f;
        #pragma unroll
        for (int r = 0; r < 8; ++r) {
            int m = mBase + r + 8 * half;
            float v = acc1[r] + bv;
            if (HAS_RES) v += res[(size_t)m * Nc + n];
            if (RELU)    v = fmaxf(v, 0.f);
            Cout[(size_t)m * Nc + n] = v;
        }
    }
}

// ---------------- attention score projections: e = <h[n,h,:], a[h,:]> ----------------
__global__ void attn_scores(const float* __restrict__ h, const float* __restrict__ a_s,
                            const float* __restrict__ a_d, float* __restrict__ esrc,
                            float* __restrict__ edst, long long n)
{
    long long i = (long long)blockIdx.x * blockDim.x + threadIdx.x;  // i = node*H + head
    if (i >= n) return;
    int hh = (int)(i & (H_ - 1));
    const float* hv = h + i * D_;          // C = H*D so h + i*32 is row (node, head)
    float s1 = 0.f, s2 = 0.f;
    #pragma unroll
    for (int d = 0; d < D_; ++d) {
        float x = hv[d];
        s1 += x * a_s[hh * D_ + d];
        s2 += x * a_d[hh * D_ + d];
    }
    esrc[i] = s1;
    edst[i] = s2;
}

// ---------------- edge pass 1: leaky-relu logits + segment max (monotone bit trick) ----
__global__ void edge_logits_max(const int* __restrict__ src, const int* __restrict__ dst,
                                const float* __restrict__ esrc, const float* __restrict__ edst,
                                float* __restrict__ logits, unsigned* __restrict__ mkey)
{
    long long e = (long long)blockIdx.x * blockDim.x + threadIdx.x;
    if (e >= E_) return;
    int s = src[e], d = dst[e];
    #pragma unroll
    for (int hh = 0; hh < H_; ++hh) {
        float lg = esrc[(size_t)s * H_ + hh] + edst[(size_t)d * H_ + hh];
        lg = lg > 0.f ? lg : NEG_SLOPE * lg;
        logits[e * H_ + hh] = lg;
        unsigned u = __float_as_uint(lg);
        u = (u & 0x80000000u) ? ~u : (u | 0x80000000u);
        atomicMax(&mkey[(size_t)d * H_ + hh], u);
    }
}

// ---------------- decode segment max; empty/non-finite -> 0 ----------------
__global__ void decode_max(const unsigned* __restrict__ mkey, float* __restrict__ mval,
                           long long n)
{
    long long i = (long long)blockIdx.x * blockDim.x + threadIdx.x;
    if (i >= n) return;
    unsigned u = mkey[i];
    float f = (u & 0x80000000u) ? __uint_as_float(u & 0x7FFFFFFFu) : __uint_as_float(~u);
    if (!(fabsf(f) < 3.0e38f)) f = 0.f;   // NaN or inf -> 0 (reference isfinite guard)
    mval[i] = f;
}

// ---------------- edge pass 2: ex = exp(logit - m[dst]); denom[dst] += ex --------------
__global__ void edge_exp_sum(const int* __restrict__ dst, float* __restrict__ logits,
                             const float* __restrict__ mval, float* __restrict__ denom)
{
    long long e = (long long)blockIdx.x * blockDim.x + threadIdx.x;
    if (e >= E_) return;
    int d = dst[e];
    #pragma unroll
    for (int hh = 0; hh < H_; ++hh) {
        float ex = __expf(logits[e * H_ + hh] - mval[(size_t)d * H_ + hh]);
        logits[e * H_ + hh] = ex;
        atomicAdd(&denom[(size_t)d * H_ + hh], ex);
    }
}

// ---------------- edge pass 3: msg[dst,:] += alpha * h[src,:] ----------------
__global__ void edge_scatter(const int* __restrict__ src, const int* __restrict__ dst,
                             const float* __restrict__ ex, const float* __restrict__ denom,
                             const float* __restrict__ h, float* __restrict__ msg)
{
    int e = blockIdx.x;
    int c = threadIdx.x;          // feature 0..255
    int s = src[e], d = dst[e];
    int hh = c >> 5;              // head
    float alpha = ex[(size_t)e * H_ + hh] / (denom[(size_t)d * H_ + hh] + 1e-16f);
    atomicAdd(&msg[(size_t)d * C_ + c], alpha * h[(size_t)s * C_ + c]);
}

// ---------------- elementwise relu ----------------
__global__ void relu_k(const float* __restrict__ in, float* __restrict__ out, long long n)
{
    long long i = (long long)blockIdx.x * blockDim.x + threadIdx.x;
    if (i >= n) return;
    float v = in[i];
    out[i] = v > 0.f ? v : 0.f;
}

// ---------------- k = mem@Wk, v = mem@Wv (tiny: 10x128 @ 128x256) ----------------
__global__ void kv_proj(const float* __restrict__ mem, const float* __restrict__ Wk,
                        const float* __restrict__ Wv, float* __restrict__ kbuf,
                        float* __restrict__ vbuf)
{
    int row = blockIdx.x;      // 0..9
    int n   = threadIdx.x;     // 0..255
    float sk = 0.f, sv = 0.f;
    for (int kk = 0; kk < MD_; ++kk) {
        float mv = mem[(size_t)row * MD_ + kk];
        sk += mv * Wk[(size_t)kk * C_ + n];
        sv += mv * Wv[(size_t)kk * C_ + n];
    }
    kbuf[(size_t)row * C_ + n] = sk;
    vbuf[(size_t)row * C_ + n] = sv;
}

// ---------------- fused memory cross-attention + residual + relu ----------------
__global__ void cross_attn(const float* __restrict__ q, const float* __restrict__ kbuf,
                           const float* __restrict__ vbuf, const float* __restrict__ node_mid,
                           const float* __restrict__ node_in, float* __restrict__ node_out,
                           int n_nodes)
{
    __shared__ float ks[MS_ * C_];
    __shared__ float vs[MS_ * C_];
    for (int i = threadIdx.x; i < MS_ * C_; i += blockDim.x) {
        ks[i] = kbuf[i];
        vs[i] = vbuf[i];
    }
    __syncthreads();
    int n = blockIdx.x * blockDim.x + threadIdx.x;
    if (n >= n_nodes) return;

    float s[MS_];
    #pragma unroll
    for (int m = 0; m < MS_; ++m) s[m] = 0.f;

    const float* qr = q + (size_t)n * C_;
    for (int c = 0; c < C_; ++c) {
        float qc = qr[c];
        #pragma unroll
        for (int m = 0; m < MS_; ++m) s[m] += qc * ks[m * C_ + c];
    }
    const float scale = 0.0625f;  // 1/sqrt(256)
    float mx = -3.0e38f;
    #pragma unroll
    for (int m = 0; m < MS_; ++m) { s[m] *= scale; mx = s[m] > mx ? s[m] : mx; }
    float sum = 0.f;
    #pragma unroll
    for (int m = 0; m < MS_; ++m) { s[m] = __expf(s[m] - mx); sum += s[m]; }
    float inv = 1.f / sum;

    for (int c = 0; c < C_; ++c) {
        float acc = 0.f;
        #pragma unroll
        for (int m = 0; m < MS_; ++m) acc += s[m] * vs[m * C_ + c];
        float r = node_mid[(size_t)n * C_ + c] + acc * inv + node_in[(size_t)n * C_ + c];
        node_out[(size_t)n * C_ + c] = r > 0.f ? r : 0.f;
    }
}

// ---------------- column mean over nodes ----------------
__global__ void col_mean(const float* __restrict__ node, float* __restrict__ pooled,
                         int n_nodes, int cols)
{
    int c = blockIdx.x;
    float s = 0.f;
    for (int n = threadIdx.x; n < n_nodes; n += blockDim.x)
        s += node[(size_t)n * cols + c];
    __shared__ float red[256];
    red[threadIdx.x] = s;
    __syncthreads();
    for (int off = 128; off > 0; off >>= 1) {
        if (threadIdx.x < off) red[threadIdx.x] += red[threadIdx.x + off];
        __syncthreads();
    }
    if (threadIdx.x == 0) pooled[c] = red[0] / (float)n_nodes;
}

// ---------------- memory-bank mean (10 rows of 128) ----------------
__global__ void mem_mean(const float* __restrict__ gmem, float* __restrict__ mp)
{
    int c = threadIdx.x;  // 0..127
    if (c >= MD_) return;
    float s = 0.f;
    #pragma unroll
    for (int m = 0; m < MS_; ++m) s += gmem[(size_t)m * MD_ + c];
    mp[c] = s / (float)MS_;
}

// ---------------- final classifier: out = [pooled; mp] @ Wc + bc ----------------
__global__ void final_out(const float* __restrict__ pooled, const float* __restrict__ mp,
                          const float* __restrict__ Wc, const float* __restrict__ bc,
                          float* __restrict__ out)
{
    int nc = threadIdx.x;
    if (nc < NC_) {
        float s = bc[nc];
        for (int i = 0; i < C_; ++i)  s += pooled[i] * Wc[(size_t)i * NC_ + nc];
        for (int j = 0; j < MD_; ++j) s += mp[j] * Wc[(size_t)(C_ + j) * NC_ + nc];
        out[nc] = s;
    }
}

// ---------------- host orchestration ----------------
extern "C" void kernel_launch(void* const* d_in, const int* in_sizes, int n_in,
                              void* d_out, int out_size, void* d_ws, size_t ws_size,
                              hipStream_t stream)
{
    const float* x_init = (const float*)d_in[0];
    const int*   eidx   = (const int*)d_in[1];
    const float* temb   = (const float*)d_in[2];
    const float* Wi     = (const float*)d_in[3];
    const float* bi     = (const float*)d_in[4];
    const float* Wg     = (const float*)d_in[5];
    const float* a_src  = (const float*)d_in[6];
    const float* a_dst  = (const float*)d_in[7];
    const float* Wt     = (const float*)d_in[8];
    const float* Wq     = (const float*)d_in[9];
    const float* Wk     = (const float*)d_in[10];
    const float* Wv     = (const float*)d_in[11];
    const float* gmem   = (const float*)d_in[12];
    const float* Wc     = (const float*)d_in[13];
    const float* bc     = (const float*)d_in[14];

    const int* src = eidx;          // edge_index[0]
    const int* dst = eidx + E_;     // edge_index[1]

    // bump allocator over d_ws (recomputed every call; no static state)
    char*  base = (char*)d_ws;
    size_t off  = 0;
    auto balloc = [&](size_t bytes) -> void* {
        off = (off + 255) & ~(size_t)255;
        void* p = base + off;
        off += bytes;
        return p;
    };
    const size_t NCf = (size_t)N_ * C_;
    float*    nodeA  = (float*)balloc(NCf * 4);
    float*    nodeB  = (float*)balloc(NCf * 4);
    float*    bufX   = (float*)balloc(NCf * 4);   // xt / node_mid
    float*    bufH   = (float*)balloc(NCf * 4);   // h  / q
    float*    bufM   = (float*)balloc(NCf * 4);   // msg accumulator
    float*    esrc   = (float*)balloc((size_t)N_ * H_ * 4);
    float*    edstb  = (float*)balloc((size_t)N_ * H_ * 4);
    unsigned* mkey   = (unsigned*)balloc((size_t)N_ * H_ * 4);
    float*    mval   = (float*)balloc((size_t)N_ * H_ * 4);
    float*    denom  = (float*)balloc((size_t)N_ * H_ * 4);
    float*    exbuf  = (float*)balloc((size_t)E_ * H_ * 4);  // logits -> exp
    float*    kbuf   = (float*)balloc((size_t)MS_ * C_ * 4);
    float*    vbuf   = (float*)balloc((size_t)MS_ * C_ * 4);
    float*    pooled = (float*)balloc(C_ * 4);
    float*    mpool  = (float*)balloc(MD_ * 4);
    // pre-packed bf16 weight buffers (dwords = K*Nc/2 each)
    unsigned* pWi = (unsigned*)balloc((size_t)FEAT * C_ / 2 * 4);
    unsigned* pWt[L_];
    unsigned* pWg[L_];
    unsigned* pWq[L_];
    for (int l = 0; l < L_; ++l) {
        pWt[l] = (unsigned*)balloc((size_t)T_ * C_ / 2 * 4);
        pWg[l] = (unsigned*)balloc((size_t)C_ * C_ / 2 * 4);
        pWq[l] = (unsigned*)balloc((size_t)C_ * C_ / 2 * 4);
    }

    const dim3 blk256(256);
    const int  gM16  = N_ / 16;                       // 1250 row tiles
    const int  gNH   = (N_ * H_ + 255) / 256;         // 625
    const int  gE    = (E_ + 255) / 256;              // 1250
    const int  gNC   = (int)((NCf + 255) / 256);      // 20000
    const int  gNode = (N_ + 255) / 256;              // 79

    auto pack = [&](const float* W, unsigned* Wp, int K) {
        int total = K * C_ / 2;
        pack_b_bf16<<<(total + 255) / 256, blk256, 0, stream>>>(W, Wp, K, C_);
    };
    // pack all weights up front (constant per call; stream-ordered)
    pack(Wi, pWi, FEAT);
    for (int l = 0; l < L_; ++l) {
        pack(Wt + (size_t)l * T_ * C_, pWt[l], T_);
        pack(Wg + (size_t)l * C_ * C_, pWg[l], C_);
        pack(Wq + (size_t)l * C_ * C_, pWq[l], C_);
    }

    // node = relu(x @ Wi + bi)
    wmma_gemm<true, false, true><<<gM16, blk256, 0, stream>>>(
        x_init, pWi, bi, nullptr, nodeA, N_, FEAT, C_);

    float* nodeIn  = nodeA;
    float* nodeOut = nodeB;

    for (int l = 0; l < L_; ++l) {
        const float* as_l = a_src + (size_t)l * H_ * D_;
        const float* ad_l = a_dst + (size_t)l * H_ * D_;
        const float* Wk_l = Wk + (size_t)l * MD_ * C_;
        const float* Wv_l = Wv + (size_t)l * MD_ * C_;

        // xt = temb @ Wt[l] + node   (time fusion, fused residual)
        wmma_gemm<false, true, false><<<gM16, blk256, 0, stream>>>(
            temb, pWt[l], nullptr, nodeIn, bufX, N_, T_, C_);
        // h = xt @ Wg[l]
        wmma_gemm<false, false, false><<<gM16, blk256, 0, stream>>>(
            bufX, pWg[l], nullptr, nullptr, bufH, N_, C_, C_);
        // per-(node,head) attention scalars
        attn_scores<<<gNH, blk256, 0, stream>>>(bufH, as_l, ad_l, esrc, edstb,
                                                (long long)N_ * H_);
        // segment max via ordered-uint atomicMax
        fill_u32<<<gNH, blk256, 0, stream>>>(mkey, 0u, (long long)N_ * H_);
        edge_logits_max<<<gE, blk256, 0, stream>>>(src, dst, esrc, edstb, exbuf, mkey);
        decode_max<<<gNH, blk256, 0, stream>>>(mkey, mval, (long long)N_ * H_);
        // segment sum of exp
        fill_f32<<<gNH, blk256, 0, stream>>>(denom, 0.f, (long long)N_ * H_);
        edge_exp_sum<<<gE, blk256, 0, stream>>>(dst, exbuf, mval, denom);
        // weighted message scatter
        fill_f32<<<gNC, blk256, 0, stream>>>(bufM, 0.f, (long long)NCf);
        edge_scatter<<<E_, blk256, 0, stream>>>(src, dst, exbuf, denom, bufH, bufM);
        // node_mid = relu(msg)   (reuse bufX)
        relu_k<<<gNC, blk256, 0, stream>>>(bufM, bufX, (long long)NCf);
        // q = node_mid @ Wq[l]   (reuse bufH)
        wmma_gemm<false, false, false><<<gM16, blk256, 0, stream>>>(
            bufX, pWq[l], nullptr, nullptr, bufH, N_, C_, C_);
        // k/v projections of memory bank
        kv_proj<<<MS_, blk256, 0, stream>>>(gmem, Wk_l, Wv_l, kbuf, vbuf);
        // fused cross-attention + residual + relu
        cross_attn<<<gNode, blk256, 0, stream>>>(bufH, kbuf, vbuf, bufX, nodeIn,
                                                 nodeOut, N_);
        // ping-pong
        float* t = nodeIn; nodeIn = nodeOut; nodeOut = t;
    }

    // pooled = mean(node, axis=0); mem_pooled = mean(global_memory, axis=0)
    col_mean<<<C_, blk256, 0, stream>>>(nodeIn, pooled, N_, C_);
    mem_mean<<<1, dim3(128), 0, stream>>>(gmem, mpool);
    final_out<<<1, dim3(32), 0, stream>>>(pooled, mpool, Wc, bc, (float*)d_out);
}
